// MoELayer_41575283425890
// MI455X (gfx1250) — matrix operand order repaired and checked by reading, not compiled, not run on hardware
//
#include <hip/hip_runtime.h>
#include <hip/hip_bf16.h>

typedef __attribute__((ext_vector_type(16))) __bf16 v16bf;
typedef __attribute__((ext_vector_type(8)))  float  v8f;

#define T_TOK  4096
#define DDIM   1024
#define FDIM   4096
#define NEXP   8
#define FC     256       // F chunk staged in LDS per iteration
#define TILE_M 32        // tokens per workgroup

__device__ __forceinline__ float gelu_tanh(float x) {
    float x3 = x * x * x;
    return 0.5f * x * (1.0f + tanhf(0.7978845608028654f * (x + 0.044715f * x3)));
}

__global__ void moe_init(int* cnt, float* usage) {
    int i = threadIdx.x;
    if (i < NEXP) { cnt[i] = 0; usage[i] = 0.0f; }
}

// One wave32 per token: router logits, softmax, top-2 gates, bucket scatter,
// usage accumulation, and zero-init of this token's output row.
__global__ void moe_router(const float* __restrict__ x, const float* __restrict__ Wr,
                           float* __restrict__ out, float* __restrict__ usage,
                           int* __restrict__ cnt, int* __restrict__ btok,
                           float* __restrict__ bgate) {
    int t    = (blockIdx.x * blockDim.x + threadIdx.x) >> 5;
    int lane = threadIdx.x & 31;
    if (t >= T_TOK) return;

    // Zero this token's output row (FFN kernel accumulates with f32 atomics).
    float4 z = {0.0f, 0.0f, 0.0f, 0.0f};
    float4* orow = (float4*)(out + (size_t)t * DDIM);
    #pragma unroll
    for (int i = 0; i < DDIM / 4 / 32; ++i) orow[lane + i * 32] = z;

    float acc[NEXP];
    #pragma unroll
    for (int e = 0; e < NEXP; ++e) acc[e] = 0.0f;
    const float* xr = x + (size_t)t * DDIM;
    for (int d = lane; d < DDIM; d += 32) {
        float xv = xr[d];
        #pragma unroll
        for (int e = 0; e < NEXP; ++e) acc[e] += xv * Wr[e * DDIM + d];
    }
    #pragma unroll
    for (int e = 0; e < NEXP; ++e) {
        #pragma unroll
        for (int o = 16; o > 0; o >>= 1) acc[e] += __shfl_xor(acc[e], o, 32);
    }

    if (lane == 0) {
        float mx = acc[0];
        #pragma unroll
        for (int e = 1; e < NEXP; ++e) mx = fmaxf(mx, acc[e]);
        float p[NEXP]; float s = 0.0f;
        #pragma unroll
        for (int e = 0; e < NEXP; ++e) { p[e] = __expf(acc[e] - mx); s += p[e]; }
        float inv = 1.0f / s;
        #pragma unroll
        for (int e = 0; e < NEXP; ++e) atomicAdd(&usage[e], p[e] * inv);

        // top-2 (first index wins ties, matching lax.top_k)
        int i0 = 0;
        #pragma unroll
        for (int e = 1; e < NEXP; ++e) if (p[e] > p[i0]) i0 = e;
        int i1 = (i0 == 0) ? 1 : 0;
        #pragma unroll
        for (int e = 0; e < NEXP; ++e) if (e != i0 && p[e] > p[i1]) i1 = e;
        float gs = 1.0f / (p[i0] + p[i1]);
        int p0 = atomicAdd(&cnt[i0], 1);
        btok[i0 * T_TOK + p0] = t;  bgate[i0 * T_TOK + p0] = p[i0] * gs;
        int p1 = atomicAdd(&cnt[i1], 1);
        btok[i1 * T_TOK + p1] = t;  bgate[i1 * T_TOK + p1] = p[i1] * gs;
    }
}

__global__ void f32_to_bf16(const float* __restrict__ in, __bf16* __restrict__ o, size_t n) {
    size_t i = blockIdx.x * (size_t)blockDim.x + threadIdx.x;
    size_t stride = (size_t)gridDim.x * blockDim.x;
    for (; i < n; i += stride) o[i] = (__bf16)in[i];
}

union Frag { v16bf v; uint4 q[2]; };

// Per workgroup: expert e, 32-token tile, 8 waves.
// Loop F in chunks of FC: phase1 h=GELU(X*W1^T+b1) chunk -> LDS (bf16),
// phase2 Y += h * W2^T with persistent f32 accumulators. Gate-weighted
// atomic-add epilogue into output rows. Register-lean (<256 VGPR) so 2+
// waves/SIMD can co-hide load latency; weight lines are prefetched ~2
// k-steps ahead (global_prefetch_b8, no VGPR/counter cost).
__global__ __launch_bounds__(256, 1)
void moe_ffn(const __bf16* __restrict__ xb, const __bf16* __restrict__ W1b,
             const __bf16* __restrict__ W2b,
             const float* __restrict__ b1, const float* __restrict__ b2,
             const int* __restrict__ cnt, const int* __restrict__ btok,
             const float* __restrict__ bgate, float* __restrict__ out) {
    constexpr int HPAD = FC + 8;                 // row pad: keep 16B align, spread banks
    __shared__ __bf16 hbuf[TILE_M][HPAD];        // ~16.9 KB
    __shared__ int    toks[TILE_M];
    __shared__ float  gts[TILE_M];

    const int e   = blockIdx.y;
    const int t32 = blockIdx.x;
    const int ce  = cnt[e];
    if (t32 * TILE_M >= ce) return;

    const int tid = threadIdx.x;
    if (tid < TILE_M) {
        int pos = t32 * TILE_M + tid;
        if (pos < ce) { toks[tid] = btok[e * T_TOK + pos]; gts[tid] = bgate[e * T_TOK + pos]; }
        else          { toks[tid] = 0;                     gts[tid] = 0.0f; }
    }
    __syncthreads();

    const int w    = tid >> 5;       // wave 0..7
    const int lane = tid & 31;
    const int lo16 = lane & 15;
    const int hi   = (lane >> 4) & 1; // K-half select per ISA 16-bit A/B layout

    // Persistent Y accumulators: this wave owns D columns [w*128, w*128+128)
    v8f yacc[2][8];
    #pragma unroll
    for (int nt = 0; nt < 8; ++nt) {
        float bv = b2[e * DDIM + w * 128 + nt * 16 + lo16];
        #pragma unroll
        for (int v = 0; v < 8; ++v) { yacc[0][nt][v] = bv; yacc[1][nt][v] = bv; }
    }

    const __bf16* W1e = W1b + (size_t)e * FDIM * DDIM;
    const __bf16* W2e = W2b + (size_t)e * DDIM * FDIM;
    // A rows (tokens) for the two M-tiles; per-lane base honors K-half striping
    const __bf16* a0row = xb + (size_t)toks[lo16]      * DDIM + hi * 8;
    const __bf16* a1row = xb + (size_t)toks[16 + lo16] * DDIM + hi * 8;

    for (int fc = 0; fc < FDIM; fc += FC) {
        // ---------------- phase 1: h chunk (this wave: 32 F-cols) ----------------
        v8f hacc[2][2];
        #pragma unroll
        for (int ntF = 0; ntF < 2; ++ntF) {
            float bv = b1[e * FDIM + fc + w * 32 + ntF * 16 + lo16];
            #pragma unroll
            for (int v = 0; v < 8; ++v) { hacc[0][ntF][v] = bv; hacc[1][ntF][v] = bv; }
        }
        const __bf16* b0row = W1e + (size_t)(fc + w * 32 + lo16)      * DDIM + hi * 8;
        const __bf16* b1row = W1e + (size_t)(fc + w * 32 + 16 + lo16) * DDIM + hi * 8;

        for (int k0 = 0; k0 < DDIM; k0 += 32) {
            // Prefetch W1 lines two k-steps ahead (speculative; OOB dropped).
            __builtin_prefetch(b0row + k0 + 64, 0, 3);
            __builtin_prefetch(b1row + k0 + 64, 0, 3);
            Frag fa0, fa1, fb0, fb1;
            fa0.q[0] = *(const uint4*)(a0row + k0);
            fa0.q[1] = *(const uint4*)(a0row + k0 + 16);
            fa1.q[0] = *(const uint4*)(a1row + k0);
            fa1.q[1] = *(const uint4*)(a1row + k0 + 16);
            fb0.q[0] = *(const uint4*)(b0row + k0);
            fb0.q[1] = *(const uint4*)(b0row + k0 + 16);
            fb1.q[0] = *(const uint4*)(b1row + k0);
            fb1.q[1] = *(const uint4*)(b1row + k0 + 16);
            hacc[0][0] = __builtin_amdgcn_wmma_f32_16x16x32_bf16(false, fa0.v, false, fb0.v, (short)0, hacc[0][0], false, false);
            hacc[1][0] = __builtin_amdgcn_wmma_f32_16x16x32_bf16(false, fa1.v, false, fb0.v, (short)0, hacc[1][0], false, false);
            hacc[0][1] = __builtin_amdgcn_wmma_f32_16x16x32_bf16(false, fa0.v, false, fb1.v, (short)0, hacc[0][1], false, false);
            hacc[1][1] = __builtin_amdgcn_wmma_f32_16x16x32_bf16(false, fa1.v, false, fb1.v, (short)0, hacc[1][1], false, false);
        }

        // GELU + stage bf16 chunk in LDS (C/D layout: M = hi*8 + v, N = lo16)
        #pragma unroll
        for (int ntF = 0; ntF < 2; ++ntF) {
            int c = w * 32 + ntF * 16 + lo16;
            #pragma unroll
            for (int mt = 0; mt < 2; ++mt) {
                #pragma unroll
                for (int v = 0; v < 8; ++v) {
                    hbuf[mt * 16 + hi * 8 + v][c] = (__bf16)gelu_tanh(hacc[mt][ntF][v]);
                }
            }
        }
        __syncthreads();

        // ---------------- phase 2: Y += h_chunk * W2^T ----------------
        const __bf16* hr0 = &hbuf[lo16][hi * 8];
        const __bf16* hr1 = &hbuf[16 + lo16][hi * 8];
        for (int k0 = 0; k0 < FC; k0 += 32) {
            Frag ha0, ha1;
            ha0.q[0] = *(const uint4*)(hr0 + k0);
            ha0.q[1] = *(const uint4*)(hr0 + k0 + 16);
            ha1.q[0] = *(const uint4*)(hr1 + k0);
            ha1.q[1] = *(const uint4*)(hr1 + k0 + 16);
            #pragma unroll
            for (int nt = 0; nt < 8; ++nt) {
                const __bf16* brow = W2e + (size_t)(w * 128 + nt * 16 + lo16) * FDIM + fc + hi * 8;
                // Prefetch this row two k-steps ahead.
                __builtin_prefetch(brow + k0 + 64, 0, 3);
                Frag wb;
                wb.q[0] = *(const uint4*)(brow + k0);
                wb.q[1] = *(const uint4*)(brow + k0 + 16);
                yacc[0][nt] = __builtin_amdgcn_wmma_f32_16x16x32_bf16(false, ha0.v, false, wb.v, (short)0, yacc[0][nt], false, false);
                yacc[1][nt] = __builtin_amdgcn_wmma_f32_16x16x32_bf16(false, ha1.v, false, wb.v, (short)0, yacc[1][nt], false, false);
            }
        }
        __syncthreads();   // before next chunk overwrites hbuf
    }

    // Epilogue: gate-weighted accumulation into output rows.
    #pragma unroll
    for (int mt = 0; mt < 2; ++mt) {
        #pragma unroll
        for (int v = 0; v < 8; ++v) {
            int m = mt * 16 + hi * 8 + v;
            float g = gts[m];
            if (g != 0.0f) {
                float* orow = out + (size_t)toks[m] * DDIM;
                #pragma unroll
                for (int nt = 0; nt < 8; ++nt) {
                    atomicAdd(&orow[w * 128 + nt * 16 + lo16], g * yacc[mt][nt][v]);
                }
            }
        }
    }
}

__global__ void moe_aux(const float* __restrict__ usage, float* __restrict__ out_aux) {
    if (threadIdx.x == 0) {
        float s = 0.0f;
        #pragma unroll
        for (int e = 0; e < NEXP; ++e) {
            float u = usage[e] / (float)T_TOK - 1.0f / (float)NEXP;
            s += u * u;
        }
        *out_aux = 0.01f * s;
    }
}

extern "C" void kernel_launch(void* const* d_in, const int* in_sizes, int n_in,
                              void* d_out, int out_size, void* d_ws, size_t ws_size,
                              hipStream_t stream) {
    (void)in_sizes; (void)n_in; (void)out_size; (void)ws_size;
    const float* x  = (const float*)d_in[0];
    const float* Wr = (const float*)d_in[1];
    const float* W1 = (const float*)d_in[2];
    const float* b1 = (const float*)d_in[3];
    const float* W2 = (const float*)d_in[4];
    const float* b2 = (const float*)d_in[5];
    float* out = (float*)d_out;

    // Workspace layout (~143 MB)
    char* ws = (char*)d_ws;
    size_t xb_bytes = (size_t)T_TOK * DDIM * 2;            // 8 MB
    size_t w_bytes  = (size_t)NEXP * FDIM * DDIM * 2;      // 64 MB each
    __bf16* xb   = (__bf16*)(ws);
    __bf16* W1b  = (__bf16*)(ws + xb_bytes);
    __bf16* W2b  = (__bf16*)(ws + xb_bytes + w_bytes);
    char*   meta = ws + xb_bytes + 2 * w_bytes;
    int*    cnt   = (int*)(meta);
    float*  usage = (float*)(meta + 32);
    int*    btok  = (int*)(meta + 64);
    float*  bgate = (float*)(meta + 64 + (size_t)NEXP * T_TOK * 4);

    moe_init<<<1, 64, 0, stream>>>(cnt, usage);
    moe_router<<<T_TOK / 8, 256, 0, stream>>>(x, Wr, out, usage, cnt, btok, bgate);
    f32_to_bf16<<<2048, 256, 0, stream>>>(x,  xb,  (size_t)T_TOK * DDIM);
    f32_to_bf16<<<4096, 256, 0, stream>>>(W1, W1b, (size_t)NEXP * FDIM * DDIM);
    f32_to_bf16<<<4096, 256, 0, stream>>>(W2, W2b, (size_t)NEXP * DDIM * FDIM);
    dim3 grid(T_TOK / TILE_M, NEXP);
    moe_ffn<<<grid, 256, 0, stream>>>(xb, W1b, W2b, b1, b2, cnt, btok, bgate, out);
    moe_aux<<<1, 32, 0, stream>>>(usage, out + (size_t)T_TOK * DDIM);
}
